// EdgeDecoder_75797582840078
// MI455X (gfx1250) — matrix operand order repaired and checked by reading, not compiled, not run on hardware
//
#include <hip/hip_runtime.h>

#define HIDDEN   256
#define K_DIM    512
#define N_EDGES  300000
#define M_TILE   48    // 3 M-subtiles of 16 -> each B fragment reused 3x; 48 | 300000
#define M_SUB    3
#define A_STRIDE 520   // 512 + 8 bf16 elems: rows stay 16B-aligned, rotate 4 LDS banks/row

typedef __attribute__((ext_vector_type(16))) __bf16 v16bf;
typedef __attribute__((ext_vector_type(8)))  float  v8f;

union Frag16 { uint4 q[2]; v16bf v; };

__device__ __forceinline__ unsigned short f32_to_bf16_rn(float x) {
    unsigned u = __float_as_uint(x);
    u += 0x7FFFu + ((u >> 16) & 1u);     // round-to-nearest-even
    return (unsigned short)(u >> 16);
}
__device__ __forceinline__ float bf16_to_f32(unsigned short h) {
    return __uint_as_float(((unsigned)h) << 16);
}

// One-shot: W1 (512x256 f32, row=k) -> transposed [N][K] bf16 hi/lo arrays in d_ws.
__global__ __launch_bounds__(256) void prep_w1(const float* __restrict__ W1,
                                               unsigned short* __restrict__ Bhi,
                                               unsigned short* __restrict__ Blo) {
    int flat = blockIdx.x * 256 + threadIdx.x;   // 131072 elements total
    int k = flat >> 8;
    int n = flat & 255;
    float x = W1[flat];
    unsigned short hi = f32_to_bf16_rn(x);
    unsigned short lo = f32_to_bf16_rn(x - bf16_to_f32(hi));
    Bhi[n * K_DIM + k] = hi;
    Blo[n * K_DIM + k] = lo;
}

__global__ __launch_bounds__(256) void edge_decoder(
    const float* __restrict__ emb, const int* __restrict__ eidx,
    const float* __restrict__ b1, const float* __restrict__ W2,
    const float* __restrict__ b2,
    const unsigned short* __restrict__ Bhi, const unsigned short* __restrict__ Blo,
    float* __restrict__ out) {

    __shared__ __align__(16) unsigned short sAhi[M_TILE * A_STRIDE];  // 49,920 B
    __shared__ __align__(16) unsigned short sAlo[M_TILE * A_STRIDE];  // 49,920 B
    __shared__ int   s_src[M_TILE];
    __shared__ int   s_trg[M_TILE];
    __shared__ float s_part[M_TILE];

    const int tid   = threadIdx.x;
    const int ebase = blockIdx.x * M_TILE;

    if (tid < M_TILE) {
        s_src[tid]  = eidx[ebase + tid];
        s_trg[tid]  = eidx[N_EDGES + ebase + tid];
        s_part[tid] = 0.0f;
    }
    __syncthreads();

    // Gather 48 src + 48 trg embeddings (coalesced: node uniform per chunk),
    // split f32 -> bf16 hi/lo into LDS A (row = edge, k = [src(256) | trg(256)]).
    #pragma unroll 4
    for (int c = 0; c < (M_TILE * K_DIM) / 256; ++c) {
        int flat = c * 256 + tid;
        int i = flat >> 9;          // edge row 0..47
        int k = flat & 511;
        int node = (k < HIDDEN) ? s_src[i] : s_trg[i];
        float x = emb[node * HIDDEN + (k & (HIDDEN - 1))];
        unsigned short hi = f32_to_bf16_rn(x);
        unsigned short lo = f32_to_bf16_rn(x - bf16_to_f32(hi));
        sAhi[i * A_STRIDE + k] = hi;
        sAlo[i * A_STRIDE + k] = lo;
    }
    __syncthreads();

    const int lane    = tid & 31;
    const int wave    = tid >> 5;
    const int hi_half = lane >> 4;   // lane 16..31
    const int lrow    = lane & 15;

    // 3 M-subtiles x 2 N-tiles of accumulators
    v8f acc[M_SUB][2];
    #pragma unroll
    for (int m = 0; m < M_SUB; ++m) { acc[m][0] = (v8f){}; acc[m][1] = (v8f){}; }

    const int n0 = (2 * wave + 0) * 16 + lrow;   // B-lane column (global n)
    const int n1 = (2 * wave + 1) * 16 + lrow;

    const int a_koff = hi_half ? 8  : 0;   // 16-bit A 16x32 lane layout
    const int b_koff = hi_half ? 8  : 0;   // 16-bit B 32x16 lane layout

    #pragma unroll 2
    for (int ks = 0; ks < K_DIM / 32; ++ks) {
        const int k0 = ks * 32;
        Frag16 ah[M_SUB], al[M_SUB], bh0, bl0, bh1, bl1;

        // A fragments: rows lrow + 16*m, hi & lo
        #pragma unroll
        for (int m = 0; m < M_SUB; ++m) {
            const unsigned short* pa = &sAhi[(lrow + 16 * m) * A_STRIDE + k0 + a_koff];
            ah[m].q[0] = *(const uint4*)(pa); ah[m].q[1] = *(const uint4*)(pa + 16);
            const unsigned short* pl = &sAlo[(lrow + 16 * m) * A_STRIDE + k0 + a_koff];
            al[m].q[0] = *(const uint4*)(pl); al[m].q[1] = *(const uint4*)(pl + 16);
        }

        // B fragments: 2 n-tiles, hi & lo (each reused by all 3 M-subtiles)
        const uint4* pb;
        pb = (const uint4*)(Bhi + n0 * K_DIM + k0 + b_koff); bh0.q[0] = pb[0]; bh0.q[1] = pb[2];
        pb = (const uint4*)(Blo + n0 * K_DIM + k0 + b_koff); bl0.q[0] = pb[0]; bl0.q[1] = pb[2];
        pb = (const uint4*)(Bhi + n1 * K_DIM + k0 + b_koff); bh1.q[0] = pb[0]; bh1.q[1] = pb[2];
        pb = (const uint4*)(Blo + n1 * K_DIM + k0 + b_koff); bl1.q[0] = pb[0]; bl1.q[1] = pb[2];

        // (A_hi+A_lo)(B_hi+B_lo) ~= hi*hi + lo*hi + hi*lo  (f32 accumulate)
        #pragma unroll
        for (int m = 0; m < M_SUB; ++m) {
            acc[m][0] = __builtin_amdgcn_wmma_f32_16x16x32_bf16(false, ah[m].v, false, bh0.v, (short)0, acc[m][0], false, false);
            acc[m][0] = __builtin_amdgcn_wmma_f32_16x16x32_bf16(false, al[m].v, false, bh0.v, (short)0, acc[m][0], false, false);
            acc[m][0] = __builtin_amdgcn_wmma_f32_16x16x32_bf16(false, ah[m].v, false, bl0.v, (short)0, acc[m][0], false, false);
            acc[m][1] = __builtin_amdgcn_wmma_f32_16x16x32_bf16(false, ah[m].v, false, bh1.v, (short)0, acc[m][1], false, false);
            acc[m][1] = __builtin_amdgcn_wmma_f32_16x16x32_bf16(false, al[m].v, false, bh1.v, (short)0, acc[m][1], false, false);
            acc[m][1] = __builtin_amdgcn_wmma_f32_16x16x32_bf16(false, ah[m].v, false, bl1.v, (short)0, acc[m][1], false, false);
        }
    }

    // Epilogue: relu(h + b1) * W2, reduce over 256 columns per edge.
    const float b1v0 = b1[n0], w2v0 = W2[n0];
    const float b1v1 = b1[n1], w2v1 = W2[n1];
    #pragma unroll
    for (int m = 0; m < M_SUB; ++m) {
        #pragma unroll
        for (int r = 0; r < 8; ++r) {
            const int row = r + (hi_half ? 8 : 0);   // C layout: VGPR r -> M=r / r+8
            float t = fmaxf(acc[m][0][r] + b1v0, 0.0f) * w2v0
                    + fmaxf(acc[m][1][r] + b1v1, 0.0f) * w2v1;
            #pragma unroll
            for (int x = 1; x < 16; x <<= 1) t += __shfl_xor(t, x, 32);
            if (lrow == 0) atomicAdd(&s_part[16 * m + row], t);
        }
    }
    __syncthreads();

    if (tid < M_TILE) out[ebase + tid] = s_part[tid] + b2[0];
}

extern "C" void kernel_launch(void* const* d_in, const int* in_sizes, int n_in,
                              void* d_out, int out_size, void* d_ws, size_t ws_size,
                              hipStream_t stream) {
    (void)in_sizes; (void)n_in; (void)out_size; (void)ws_size;
    const float* emb = (const float*)d_in[0];
    const int*   idx = (const int*)  d_in[1];
    const float* W1  = (const float*)d_in[2];
    const float* b1  = (const float*)d_in[3];
    const float* W2  = (const float*)d_in[4];
    const float* b2  = (const float*)d_in[5];
    float* out = (float*)d_out;

    unsigned short* Bhi = (unsigned short*)d_ws;
    unsigned short* Blo = Bhi + (size_t)K_DIM * HIDDEN;

    prep_w1<<<(K_DIM * HIDDEN) / 256, 256, 0, stream>>>(W1, Bhi, Blo);
    edge_decoder<<<N_EDGES / M_TILE, 256, 0, stream>>>(emb, idx, b1, W2, b2, Bhi, Blo, out);
}
